// HybridLongformerDetector_9131100471944
// MI455X (gfx1250) — matrix operand order repaired and checked by reading, not compiled, not run on hardware
//
#include <hip/hip_runtime.h>
#include <hip/hip_bf16.h>
#include <math.h>

namespace {

constexpr int S_LEN   = 4096;
constexpr int D_MODEL = 768;
constexpr int N_HEAD  = 12;
constexpr int D_HEAD  = 64;
constexpr int N_LAYER = 12;
constexpr int FF_DIM  = 3072;
constexpr int W_WIN   = 256;
constexpr int P_CHUNK = S_LEN / W_WIN;     // 16
constexpr int S_PAD   = S_LEN + 2 * W_WIN; // 4608 (padded key axis)
constexpr float NEG_BIG = -1.0e9f;

typedef __attribute__((ext_vector_type(16))) __bf16        v16bf;
typedef __attribute__((ext_vector_type(8)))  float         v8f;
typedef __attribute__((ext_vector_type(4)))  unsigned int  u32x4;

union FragAB { v16bf v; u32x4 u[2]; };

__device__ __forceinline__ unsigned short f2bf(float x) {
  unsigned int u = __float_as_uint(x);
  u += 0x7FFFu + ((u >> 16) & 1u);           // round-to-nearest-even
  return (unsigned short)(u >> 16);
}
__device__ __forceinline__ float bf2f(unsigned short h) {
  return __uint_as_float(((unsigned int)h) << 16);
}

__device__ __forceinline__ float gelu_tanh(float x) {
  const float c0 = 0.7978845608028654f, c1 = 0.044715f;
  return 0.5f * x * (1.0f + tanhf(c0 * (x + c1 * x * x * x)));
}

__device__ __forceinline__ float block_sum256(float v, float* red) {
  int t = threadIdx.x;
  red[t] = v; __syncthreads();
  #pragma unroll
  for (int o = 128; o > 0; o >>= 1) {
    if (t < o) red[t] += red[t + o];
    __syncthreads();
  }
  float r = red[0]; __syncthreads();
  return r;
}

__device__ __forceinline__ float block_max256(float v, float* red) {
  int t = threadIdx.x;
  red[t] = v; __syncthreads();
  #pragma unroll
  for (int o = 128; o > 0; o >>= 1) {
    if (t < o) red[t] = fmaxf(red[t], red[t + o]);
    __syncthreads();
  }
  float r = red[0]; __syncthreads();
  return r;
}

// ---------------------------------------------------------------------------
// Zero the W padding rows (front and back) of kpad
// ---------------------------------------------------------------------------
__global__ __launch_bounds__(256)
void kpad_zero_kernel(unsigned short* __restrict__ kp) {
  int i = blockIdx.x * 256 + threadIdx.x;          // over 2*W*D
  if (i >= 2 * W_WIN * D_MODEL) return;
  int j = (i < W_WIN * D_MODEL) ? i : (i - W_WIN * D_MODEL) + (S_LEN + W_WIN) * D_MODEL;
  kp[j] = 0;
}

// ---------------------------------------------------------------------------
// f32 [K,N] -> bf16 [N,K] (weight transpose, LDS tiled for coalescing)
// ---------------------------------------------------------------------------
__global__ __launch_bounds__(256)
void transpose_cvt_kernel(const float* __restrict__ src, unsigned short* __restrict__ dst,
                          int K, int N) {
  __shared__ float tile[16][17];
  int kBase = blockIdx.y * 16, nBase = blockIdx.x * 16;
  int tx = threadIdx.x & 15, ty = threadIdx.x >> 4;
  tile[ty][tx] = src[(size_t)(kBase + ty) * N + nBase + tx];
  __syncthreads();
  dst[(size_t)(nBase + ty) * K + kBase + tx] = f2bf(tile[tx][ty]);
}

// ---------------------------------------------------------------------------
// v [S,D] bf16 -> vt [H][DH][S_PAD] bf16 (per-head transpose, zero padded)
// ---------------------------------------------------------------------------
__global__ __launch_bounds__(256)
void build_vt_kernel(const unsigned short* __restrict__ v, unsigned short* __restrict__ vt) {
  __shared__ unsigned short tile[16][17];
  int rBase = blockIdx.x * 16;                 // padded key axis tile
  int h = blockIdx.y >> 2, dt = blockIdx.y & 3;
  int tx = threadIdx.x & 15, ty = threadIdx.x >> 4;
  int s = rBase + ty - W_WIN;
  tile[ty][tx] = (s >= 0 && s < S_LEN)
                   ? v[(size_t)s * D_MODEL + h * D_HEAD + dt * 16 + tx] : (unsigned short)0;
  __syncthreads();
  vt[(size_t)(h * D_HEAD + dt * 16 + ty) * S_PAD + rBase + tx] = tile[tx][ty];
}

// ---------------------------------------------------------------------------
// WMMA bf16 GEMM: C[M,N] = A[M,K] @ W[K,N] + bias, W given transposed [N,K].
// Block = 256 threads = 8 waves; block tile 128x128; wave tile 32x64 (2x4 frags).
// Epilogue: optional tanh-GELU, optional scale, f32 or bf16 store.
// ---------------------------------------------------------------------------
__global__ __launch_bounds__(256)
void gemm_bf16_wmma(const unsigned short* __restrict__ A,
                    const unsigned short* __restrict__ Bt,
                    const float* __restrict__ bias,
                    void* __restrict__ Cout,
                    int M, int N, int K, int fuse_gelu, int store_bf16, float out_scale) {
  (void)M;
  const int lane = threadIdx.x & 31;
  const int wave = threadIdx.x >> 5;
  const int lrow = lane & 15;
  const int half = lane >> 4;
  const int wr = wave >> 1;
  const int wc = wave & 1;
  const size_t mBase = (size_t)blockIdx.y * 128 + (size_t)wr * 32;
  const size_t nBase = (size_t)blockIdx.x * 128 + (size_t)wc * 64;

  v8f zero = {0.f, 0.f, 0.f, 0.f, 0.f, 0.f, 0.f, 0.f};
  v8f acc[2][4];
  #pragma unroll
  for (int i = 0; i < 2; i++)
    #pragma unroll
    for (int j = 0; j < 4; j++)
      acc[i][j] = zero;

  const unsigned short* aRow = A  + (mBase + (size_t)lrow) * (size_t)K;
  const unsigned short* bRow = Bt + (nBase + (size_t)lrow) * (size_t)K;

  for (int k0 = 0; k0 < K; k0 += 32) {
    FragAB a[2], b[4];
    #pragma unroll
    for (int i = 0; i < 2; i++) {
      const unsigned short* p = aRow + (size_t)i * 16 * K + k0 + half * 8;
      a[i].u[0] = *(const u32x4*)(const void*)(p);
      a[i].u[1] = *(const u32x4*)(const void*)(p + 16);
    }
    #pragma unroll
    for (int j = 0; j < 4; j++) {
      const unsigned short* p = bRow + (size_t)j * 16 * K + k0 + half * 8;
      b[j].u[0] = *(const u32x4*)(const void*)(p);
      b[j].u[1] = *(const u32x4*)(const void*)(p + 16);
    }
    #pragma unroll
    for (int i = 0; i < 2; i++)
      #pragma unroll
      for (int j = 0; j < 4; j++)
        acc[i][j] = __builtin_amdgcn_wmma_f32_16x16x32_bf16(
            false, a[i].v, false, b[j].v, (short)0, acc[i][j], false, false);
  }

  float* cf = (float*)Cout;
  unsigned short* ch = (unsigned short*)Cout;
  #pragma unroll
  for (int i = 0; i < 2; i++) {
    const size_t m0 = mBase + (size_t)i * 16 + (size_t)half * 8;
    #pragma unroll
    for (int j = 0; j < 4; j++) {
      const size_t n = nBase + (size_t)j * 16 + lrow;
      const float bv = bias[n];
      #pragma unroll
      for (int r = 0; r < 8; r++) {
        float val = acc[i][j][r] + bv;
        if (fuse_gelu) val = gelu_tanh(val);
        val *= out_scale;
        const size_t idx = (m0 + r) * (size_t)N + n;
        if (store_bf16) ch[idx] = f2bf(val); else cf[idx] = val;
      }
    }
  }
}

// ---------------------------------------------------------------------------
// Embedding lookup + LayerNorm; writes f32 and bf16. One block per token.
// ---------------------------------------------------------------------------
__global__ __launch_bounds__(256)
void embed_ln_kernel(const int* __restrict__ ids,
                     const float* __restrict__ wemb, const float* __restrict__ pemb,
                     const float* __restrict__ g, const float* __restrict__ b,
                     float* __restrict__ x, unsigned short* __restrict__ xbf) {
  __shared__ float red[256];
  int s = blockIdx.x, tid = threadIdx.x;
  const float* wr = wemb + (size_t)ids[s] * D_MODEL;
  const float* pr = pemb + (size_t)(s + 2) * D_MODEL;
  float vals[3];
  float ls = 0.f;
  #pragma unroll
  for (int j = 0; j < 3; j++) { int d = j * 256 + tid; vals[j] = wr[d] + pr[d]; ls += vals[j]; }
  float mean = block_sum256(ls, red) * (1.0f / D_MODEL);
  float lv = 0.f;
  #pragma unroll
  for (int j = 0; j < 3; j++) { float t = vals[j] - mean; lv += t * t; }
  float inv = rsqrtf(block_sum256(lv, red) * (1.0f / D_MODEL) + 1e-5f);
  float* xr = x + (size_t)s * D_MODEL;
  unsigned short* xh = xbf + (size_t)s * D_MODEL;
  #pragma unroll
  for (int j = 0; j < 3; j++) {
    int d = j * 256 + tid;
    float o = (vals[j] - mean) * inv * g[d] + b[d];
    xr[d] = o; xh[d] = f2bf(o);
  }
}

// ---------------------------------------------------------------------------
// out = LayerNorm(a + b); writes f32 and bf16. One block per token.
// ---------------------------------------------------------------------------
__global__ __launch_bounds__(256)
void add_ln_kernel(const float* __restrict__ a, const float* __restrict__ b,
                   const float* __restrict__ g, const float* __restrict__ beta,
                   float* __restrict__ out, unsigned short* __restrict__ outbf) {
  __shared__ float red[256];
  int s = blockIdx.x, tid = threadIdx.x;
  const float* ar = a + (size_t)s * D_MODEL;
  const float* br = b + (size_t)s * D_MODEL;
  float vals[3];
  float ls = 0.f;
  #pragma unroll
  for (int j = 0; j < 3; j++) { int d = j * 256 + tid; vals[j] = ar[d] + br[d]; ls += vals[j]; }
  float mean = block_sum256(ls, red) * (1.0f / D_MODEL);
  float lv = 0.f;
  #pragma unroll
  for (int j = 0; j < 3; j++) { float t = vals[j] - mean; lv += t * t; }
  float inv = rsqrtf(block_sum256(lv, red) * (1.0f / D_MODEL) + 1e-5f);
  float* orow = out + (size_t)s * D_MODEL;
  unsigned short* oh = outbf + (size_t)s * D_MODEL;
  #pragma unroll
  for (int j = 0; j < 3; j++) {
    int d = j * 256 + tid;
    float o = (vals[j] - mean) * inv * g[d] + beta[d];
    orow[d] = o; oh[d] = f2bf(o);
  }
}

// ---------------------------------------------------------------------------
// WMMA band attention. Grid (P_CHUNK*4, N_HEAD): chunk c = bx>>2, 64-query
// tile qt = bx&3. q tile staged via async global->LDS copy (ASYNCcnt).
// Phase 1: S = q@k3^T (WMMA) + band mask -> LDS bf16. Phase 2: softmax
// (4 lanes/row, shfl) incl. global column. Phase 3: O = P@V (WMMA) -> bf16.
// ---------------------------------------------------------------------------
__global__ __launch_bounds__(256)
void attn_band_wmma_kernel(const unsigned short* __restrict__ qbf,  // [S,D] bf16 pre-scaled
                           const unsigned short* __restrict__ kpad, // [S_PAD,D] bf16
                           const unsigned short* __restrict__ vt,   // [H][DH][S_PAD] bf16
                           const unsigned short* __restrict__ vbf,  // [S,D] bf16 (row 0 used)
                           const int* __restrict__ am,
                           unsigned short* __restrict__ out) {      // [S,D] bf16
  extern __shared__ char smem[];
  unsigned short* sbf  = (unsigned short*)smem;                     // 64*768 bf16 scores/probs
  unsigned short* qlds = (unsigned short*)(smem + 64 * 768 * 2);    // 64*64 bf16 q tile
  float* rinv = (float*)(smem + 64 * 768 * 2 + 64 * 64 * 2);        // [64] 1/l
  float* pgA  = rinv + 64;                                          // [64] p_global

  const int c = blockIdx.x >> 2, qt = blockIdx.x & 3;
  const int h = blockIdx.y;
  const int s0 = c * W_WIN + qt * 64;
  const int wave = threadIdx.x >> 5, lane = threadIdx.x & 31;
  const int lrow = lane & 15, half = lane >> 4;
  v8f zero = {0.f, 0.f, 0.f, 0.f, 0.f, 0.f, 0.f, 0.f};

  // Stage q tile (64x64 bf16, 32B/thread) via async global->LDS DMA.
  {
    int m = threadIdx.x >> 2, dch = (threadIdx.x & 3) * 16;
    const unsigned short* src = qbf + (size_t)(s0 + m) * D_MODEL + h * D_HEAD + dch;
    unsigned short* dst = qlds + m * 64 + dch;
    unsigned ldso = (unsigned)(size_t)(void*)dst;           // LDS byte address (addr[31:0])
    unsigned long long ga = (unsigned long long)(size_t)src;
    asm volatile("global_load_async_to_lds_b128 %0, %1, off"
                 :: "v"(ldso), "v"(ga) : "memory");
    asm volatile("global_load_async_to_lds_b128 %0, %1, off offset:16"
                 :: "v"(ldso), "v"(ga) : "memory");
    asm volatile("s_wait_asynccnt 0x0" ::: "memory");
  }
  __syncthreads();

  // ---- Phase 1: scores = q @ k3^T, band mask, store bf16 to LDS ----
  {
    const int wr = wave >> 2;   // 0..1 -> 32 query rows
    const int wc = wave & 3;    // 0..3 -> 192 key cols
    FragAB a[2][2];
    #pragma unroll
    for (int mi = 0; mi < 2; mi++)
      #pragma unroll
      for (int ks = 0; ks < 2; ks++) {
        const unsigned short* p = qlds + (wr * 32 + mi * 16 + lrow) * 64 + ks * 32 + half * 8;
        a[mi][ks].u[0] = *(const u32x4*)(const void*)p;
        a[mi][ks].u[1] = *(const u32x4*)(const void*)(p + 16);
      }
    for (int nf = 0; nf < 12; nf++) {
      const int n0 = wc * 192 + nf * 16;
      const int kcol = (c - 1) * W_WIN + n0 + lrow;  // absolute key position
      const unsigned short* bp = kpad + (size_t)(c * W_WIN + n0 + lrow) * D_MODEL + h * D_HEAD;
      v8f acc[2] = {zero, zero};
      #pragma unroll
      for (int ks = 0; ks < 2; ks++) {
        FragAB b;
        const unsigned short* p = bp + ks * 32 + half * 8;
        b.u[0] = *(const u32x4*)(const void*)p;
        b.u[1] = *(const u32x4*)(const void*)(p + 16);
        acc[0] = __builtin_amdgcn_wmma_f32_16x16x32_bf16(
            false, a[0][ks].v, false, b.v, (short)0, acc[0], false, false);
        acc[1] = __builtin_amdgcn_wmma_f32_16x16x32_bf16(
            false, a[1][ks].v, false, b.v, (short)0, acc[1], false, false);
      }
      bool colOk = (kcol >= 0) && (kcol < S_LEN) && (kcol != 0);
      if (colOk) colOk = (am[kcol] != 0);
      #pragma unroll
      for (int mi = 0; mi < 2; mi++) {
        #pragma unroll
        for (int r = 0; r < 8; r++) {
          int m = wr * 32 + mi * 16 + r + 8 * half;
          int rel = kcol - (s0 + m);
          bool ok = colOk && (rel >= -W_WIN) && (rel <= W_WIN);
          sbf[m * 768 + n0 + lrow] = f2bf(ok ? acc[mi][r] : NEG_BIG);
        }
      }
    }
  }
  __syncthreads();

  // ---- Phase 2: softmax over [global col, 768 band cols] ----
  {
    const int m = threadIdx.x >> 2;     // 0..63
    const int sub = threadIdx.x & 3;    // 4 lanes per row
    const int cb = sub * 192;
    float lmax = -INFINITY;
    for (int j = 0; j < 192; j++) lmax = fmaxf(lmax, bf2f(sbf[m * 768 + cb + j]));
    lmax = fmaxf(lmax, __shfl_xor(lmax, 1));
    lmax = fmaxf(lmax, __shfl_xor(lmax, 2));
    // global column score: q_row . k[s=0]  (kpad row W_WIN)
    float scg = 0.f;
    for (int d = 0; d < 64; d++)
      scg += bf2f(qlds[m * 64 + d]) * bf2f(kpad[(size_t)W_WIN * D_MODEL + h * D_HEAD + d]);
    float rowmax = fmaxf(lmax, scg);
    float lsum = 0.f;
    for (int j = 0; j < 192; j++) {
      float pr = __expf(bf2f(sbf[m * 768 + cb + j]) - rowmax);
      sbf[m * 768 + cb + j] = f2bf(pr);
      lsum += pr;
    }
    lsum += __shfl_xor(lsum, 1);
    lsum += __shfl_xor(lsum, 2);
    float pg = __expf(scg - rowmax);
    if (sub == 0) { rinv[m] = 1.0f / (lsum + pg); pgA[m] = pg; }
  }
  __syncthreads();

  // ---- Phase 3: O = P @ V, write bf16 ----
  {
    const int wr = wave >> 1;   // 0..3 -> 16 query rows
    const int wc = wave & 1;    // 0..1 -> 32 d cols
    v8f acc[2] = {zero, zero};
    for (int kk = 0; kk < 24; kk++) {
      const int k0 = kk * 32;
      FragAB a;
      const unsigned short* p = sbf + (wr * 16 + lrow) * 768 + k0 + half * 8;
      a.u[0] = *(const u32x4*)(const void*)p;
      a.u[1] = *(const u32x4*)(const void*)(p + 16);
      #pragma unroll
      for (int j = 0; j < 2; j++) {
        const int n = wc * 32 + j * 16 + lrow;  // d index
        const unsigned short* bp = vt + (size_t)(h * D_HEAD + n) * S_PAD
                                      + c * W_WIN + k0 + half * 8;
        FragAB b;
        b.u[0] = *(const u32x4*)(const void*)bp;
        b.u[1] = *(const u32x4*)(const void*)(bp + 16);
        acc[j] = __builtin_amdgcn_wmma_f32_16x16x32_bf16(
            false, a.v, false, b.v, (short)0, acc[j], false, false);
      }
    }
    #pragma unroll
    for (int j = 0; j < 2; j++) {
      const int d = wc * 32 + j * 16 + lrow;
      const float vgv = bf2f(vbf[h * D_HEAD + d]);   // v row s=0
      #pragma unroll
      for (int r = 0; r < 8; r++) {
        int m = wr * 16 + r + 8 * half;
        float val = (acc[j][r] + pgA[m] * vgv) * rinv[m];
        out[(size_t)(s0 + m) * D_MODEL + h * D_HEAD + d] = f2bf(val);
      }
    }
  }
}

// ---------------------------------------------------------------------------
// Single-row projection: out[768] = x[0,:] @ W + b  (for qg, G=1)
// ---------------------------------------------------------------------------
__global__ __launch_bounds__(256)
void rowvec_proj_kernel(const float* __restrict__ x0, const float* __restrict__ W,
                        const float* __restrict__ b, float* __restrict__ out) {
  int n = blockIdx.x * 256 + threadIdx.x;
  float a = b[n];
  for (int k = 0; k < D_MODEL; k++) a = fmaf(x0[k], W[(size_t)k * D_MODEL + n], a);
  out[n] = a;
}

// ---------------------------------------------------------------------------
// Global attention for the single global token (s=0). One block per head.
// Writes bf16 row 0 of the attention output.
// ---------------------------------------------------------------------------
__global__ __launch_bounds__(256)
void global_attn_kernel(const float* __restrict__ qg, const float* __restrict__ kg,
                        const float* __restrict__ vg, const int* __restrict__ am,
                        unsigned short* __restrict__ out) {
  __shared__ float red[256];
  __shared__ float osh[64];
  const int h = blockIdx.x, tid = threadIdx.x;
  const float scale = 0.125f;

  float qv[64];
  const float* qrow = qg + (size_t)h * D_HEAD;   // row s=0
  #pragma unroll
  for (int d = 0; d < 64; d++) qv[d] = qrow[d] * scale;

  float lm = -INFINITY;
  for (int si = tid; si < S_LEN; si += 256) {
    const float* kr = kg + (size_t)si * D_MODEL + h * D_HEAD;
    float sc = 0.f;
    #pragma unroll
    for (int d = 0; d < 64; d++) sc = fmaf(qv[d], kr[d], sc);
    if (am[si] == 0) sc += NEG_BIG;
    lm = fmaxf(lm, sc);
  }
  float gmax = block_max256(lm, red);

  if (tid < 64) osh[tid] = 0.f;
  __syncthreads();

  float ll = 0.f;
  float oloc[64];
  #pragma unroll
  for (int d = 0; d < 64; d++) oloc[d] = 0.f;
  for (int si = tid; si < S_LEN; si += 256) {
    const float* kr = kg + (size_t)si * D_MODEL + h * D_HEAD;
    const float* vr = vg + (size_t)si * D_MODEL + h * D_HEAD;
    float sc = 0.f;
    #pragma unroll
    for (int d = 0; d < 64; d++) sc = fmaf(qv[d], kr[d], sc);
    if (am[si] == 0) sc += NEG_BIG;
    float pr = __expf(sc - gmax);
    ll += pr;
    #pragma unroll
    for (int d = 0; d < 64; d++) oloc[d] = fmaf(pr, vr[d], oloc[d]);
  }
  float gsum = block_sum256(ll, red);
  #pragma unroll
  for (int d = 0; d < 64; d++) atomicAdd(&osh[d], oloc[d]);
  __syncthreads();
  if (tid < 64) out[(size_t)h * D_HEAD + tid] = f2bf(osh[tid] / gsum);  // row s=0
}

// ---------------------------------------------------------------------------
// Classifier head
// ---------------------------------------------------------------------------
__global__ __launch_bounds__(256)
void head_kernel(const float* __restrict__ x, const float* __restrict__ lf,
                 const float* __restrict__ Wl, const float* __restrict__ bl,
                 const float* __restrict__ Wc1, const float* __restrict__ bc1,
                 const float* __restrict__ Wc2, const float* __restrict__ bc2,
                 float* __restrict__ outp) {
  __shared__ float comb[D_MODEL + 64];
  __shared__ float hid[256];
  int tid = threadIdx.x;
  for (int d = tid; d < D_MODEL; d += 256) comb[d] = x[d];  // cls = x[0,:]
  if (tid < 64) {
    float a = bl[tid];
    #pragma unroll
    for (int i = 0; i < 6; i++) a = fmaf(lf[i], Wl[i * 64 + tid], a);
    comb[D_MODEL + tid] = fmaxf(a, 0.f);
  }
  __syncthreads();
  {
    float a = bc1[tid];
    for (int i = 0; i < D_MODEL + 64; i++) a = fmaf(comb[i], Wc1[(size_t)i * 256 + tid], a);
    hid[tid] = fmaxf(a, 0.f);
  }
  __syncthreads();
  if (tid < 2) {
    float a = bc2[tid];
    for (int i = 0; i < 256; i++) a = fmaf(hid[i], Wc2[i * 2 + tid], a);
    outp[tid] = a;
  }
}

} // anonymous namespace

// ---------------------------------------------------------------------------
// Host orchestration
// ---------------------------------------------------------------------------
extern "C" void kernel_launch(void* const* d_in, const int* in_sizes, int n_in,
                              void* d_out, int out_size, void* d_ws, size_t ws_size,
                              hipStream_t stream) {
  (void)in_sizes; (void)n_in; (void)out_size; (void)ws_size;

  const int*   ids   = (const int*)d_in[0];
  const int*   amask = (const int*)d_in[1];
  const float* lf    = (const float*)d_in[3];
  const float* wemb  = (const float*)d_in[4];
  const float* pemb  = (const float*)d_in[5];
  const float* eg    = (const float*)d_in[6];
  const float* eb    = (const float*)d_in[7];
  const float* Wq  = (const float*)d_in[8];   const float* bq  = (const float*)d_in[9];
  const float* Wk  = (const float*)d_in[10];  const float* bk  = (const float*)d_in[11];
  const float* Wv  = (const float*)d_in[12];  const float* bv  = (const float*)d_in[13];
  const float* Wqg = (const float*)d_in[14];  const float* bqg = (const float*)d_in[15];
  const float* Wkg = (const float*)d_in[16];  const float* bkg = (const float*)d_in[17];
  const float* Wvg = (const float*)d_in[18];  const float* bvg = (const float*)d_in[19];
  const float* Wo  = (const float*)d_in[20];  const float* bo  = (const float*)d_in[21];
  const float* l1g = (const float*)d_in[22];  const float* l1b = (const float*)d_in[23];
  const float* Wf1 = (const float*)d_in[24];  const float* bf1 = (const float*)d_in[25];
  const float* Wf2 = (const float*)d_in[26];  const float* bf2 = (const float*)d_in[27];
  const float* l2g = (const float*)d_in[28];  const float* l2b = (const float*)d_in[29];
  const float* Wl  = (const float*)d_in[30];  const float* bl  = (const float*)d_in[31];
  const float* Wc1 = (const float*)d_in[32];  const float* bc1 = (const float*)d_in[33];
  const float* Wc2 = (const float*)d_in[34];  const float* bc2 = (const float*)d_in[35];

  const size_t SD  = (size_t)S_LEN * D_MODEL;
  const size_t SF  = (size_t)S_LEN * FF_DIM;
  const size_t DD  = (size_t)D_MODEL * D_MODEL;
  const size_t DF  = (size_t)D_MODEL * FF_DIM;
  const size_t SPD = (size_t)S_PAD * D_MODEL;

  char* ws = (char*)d_ws;
  auto take = [&](size_t bytes) { char* p = ws; ws += (bytes + 255) & ~(size_t)255; return p; };
  float* x     = (float*)take(SD * 4);   // residual stream f32
  float* x2    = (float*)take(SD * 4);
  float* proj  = (float*)take(SD * 4);
  float* kgB   = (float*)take(SD * 4);
  float* vgB   = (float*)take(SD * 4);
  float* qgRow = (float*)take(D_MODEL * 4);
  unsigned short* xbf  = (unsigned short*)take(SD * 2);   // bf16 of x
  unsigned short* x2bf = (unsigned short*)take(SD * 2);   // bf16 of x2
  unsigned short* obf  = (unsigned short*)take(SD * 2);   // bf16 attention output
  unsigned short* qbf  = (unsigned short*)take(SD * 2);   // bf16 scaled q
  unsigned short* vbf  = (unsigned short*)take(SD * 2);   // bf16 v
  unsigned short* fbf  = (unsigned short*)take(SF * 2);   // bf16 gelu(FF1)
  unsigned short* wbf  = (unsigned short*)take(DF * 2);   // bf16 transposed weight
  unsigned short* kpad = (unsigned short*)take(SPD * 2);  // bf16 padded k
  unsigned short* vtB  = (unsigned short*)take(SPD * 2);  // bf16 per-head transposed v

  const int ATTN_SMEM = 64 * 768 * 2 + 64 * 64 * 2 + 2 * 64 * 4;  // 107008 B
  hipFuncSetAttribute(reinterpret_cast<const void*>(&attn_band_wmma_kernel),
                      hipFuncAttributeMaxDynamicSharedMemorySize, ATTN_SMEM);

  dim3 blk(256);
  auto tcvt = [&](const float* src, int K, int N) {
    transpose_cvt_kernel<<<dim3(N / 16, K / 16), blk, 0, stream>>>(src, wbf, K, N);
  };
  auto gemm = [&](const unsigned short* A, const float* bias, void* C,
                  int M, int N, int K, int gelu, int tobf, float scale) {
    gemm_bf16_wmma<<<dim3(N / 128, M / 128), blk, 0, stream>>>(A, wbf, bias, C,
                                                               M, N, K, gelu, tobf, scale);
  };

  embed_ln_kernel<<<dim3(S_LEN), blk, 0, stream>>>(ids, wemb, pemb, eg, eb, x, xbf);

  for (int l = 0; l < N_LAYER; l++) {
    const size_t oDD = (size_t)l * DD, oD = (size_t)l * D_MODEL;
    const size_t oDF = (size_t)l * DF, oF = (size_t)l * FF_DIM;

    // Projections (bf16 WMMA GEMMs, bf16 epilogue stores feed attention directly)
    tcvt(Wq  + oDD, D_MODEL, D_MODEL);
    gemm(xbf, bq  + oD, qbf,                      S_LEN, D_MODEL, D_MODEL, 0, 1, 0.125f);
    tcvt(Wk  + oDD, D_MODEL, D_MODEL);
    gemm(xbf, bk  + oD, kpad + (size_t)W_WIN * D_MODEL, S_LEN, D_MODEL, D_MODEL, 0, 1, 1.0f);
    tcvt(Wv  + oDD, D_MODEL, D_MODEL);
    gemm(xbf, bv  + oD, vbf,                      S_LEN, D_MODEL, D_MODEL, 0, 1, 1.0f);
    tcvt(Wkg + oDD, D_MODEL, D_MODEL);
    gemm(xbf, bkg + oD, kgB,                      S_LEN, D_MODEL, D_MODEL, 0, 0, 1.0f);
    tcvt(Wvg + oDD, D_MODEL, D_MODEL);
    gemm(xbf, bvg + oD, vgB,                      S_LEN, D_MODEL, D_MODEL, 0, 0, 1.0f);
    rowvec_proj_kernel<<<dim3(3), blk, 0, stream>>>(x, Wqg + oDD, bqg + oD, qgRow);

    // Band attention operand prep
    kpad_zero_kernel<<<dim3((2 * W_WIN * D_MODEL + 255) / 256), blk, 0, stream>>>(kpad);
    build_vt_kernel<<<dim3(S_PAD / 16, N_HEAD * 4), blk, 0, stream>>>(vbf, vtB);

    // Attention (WMMA band + scalar global token), writes bf16
    attn_band_wmma_kernel<<<dim3(P_CHUNK * 4, N_HEAD), blk, ATTN_SMEM, stream>>>(
        qbf, kpad, vtB, vbf, amask, obf);
    global_attn_kernel<<<dim3(N_HEAD), blk, 0, stream>>>(qgRow, kgB, vgB, amask, obf);

    // Output projection + residual LN (dual f32 + bf16 output)
    tcvt(Wo + oDD, D_MODEL, D_MODEL);
    gemm(obf, bo + oD, proj, S_LEN, D_MODEL, D_MODEL, 0, 0, 1.0f);
    add_ln_kernel<<<dim3(S_LEN), blk, 0, stream>>>(x, proj, l1g + oD, l1b + oD, x2, x2bf);

    // FFN: FF1 writes gelu'd bf16 directly; FF2 back to f32
    tcvt(Wf1 + oDF, D_MODEL, FF_DIM);
    gemm(x2bf, bf1 + oF, fbf, S_LEN, FF_DIM, D_MODEL, 1, 1, 1.0f);
    tcvt(Wf2 + oDF, FF_DIM, D_MODEL);
    gemm(fbf, bf2 + oD, proj, S_LEN, D_MODEL, FF_DIM, 0, 0, 1.0f);
    add_ln_kernel<<<dim3(S_LEN), blk, 0, stream>>>(x2, proj, l2g + oD, l2b + oD, x, xbf);
  }

  head_kernel<<<dim3(1), blk, 0, stream>>>(x, lf, Wl, bl, Wc1, bc1, Wc2, bc2, (float*)d_out);
}
